// DSRPN_76716705841927
// MI455X (gfx1250) — matrix operand order repaired and checked by reading, not compiled, not run on hardware
//
#include <hip/hip_runtime.h>

// ---------------------------------------------------------------------------
// Types for CDNA5 WMMA (wave32, 16x16x32 bf16 -> f32)
// ---------------------------------------------------------------------------
typedef __bf16 bf16_t;
typedef bf16_t bfvec16 __attribute__((ext_vector_type(16)));
typedef float v8f __attribute__((ext_vector_type(8)));

union Frag {
  bfvec16 v;
  uint4 q[2];
};

__device__ __forceinline__ float bf2f(unsigned short u) {
  unsigned x = ((unsigned)u) << 16;
  float f;
  __builtin_memcpy(&f, &x, 4);
  return f;
}
__device__ __forceinline__ unsigned short f2bf(float f) {
  unsigned x;
  __builtin_memcpy(&x, &f, 4);
  x = x + 0x7FFFu + ((x >> 16) & 1u);  // round-to-nearest-even
  return (unsigned short)(x >> 16);
}

// ---------------------------------------------------------------------------
// Weight convert + transpose: in [Co][Ci][3][3] fp32 ->
// out [Co][K] bf16 with K index = (kh*3+kw)*C + ci.
// With C >= 256 every 32-wide K tile then has uniform (kh,kw), killing all
// per-element div/mod in the conv gather.
// ---------------------------------------------------------------------------
__global__ void cvt_transpose_w(const float* __restrict__ in,
                                unsigned short* __restrict__ out, int C) {
  const int tot = C * C * 9;
  int i = blockIdx.x * blockDim.x + threadIdx.x;
  if (i >= tot) return;
  const int co = i / (C * 9);
  const int rem = i - co * (C * 9);
  const int ci = rem / 9;
  const int r = rem - ci * 9;
  out[(size_t)co * (C * 9) + (size_t)r * C + ci] = f2bf(in[i]);
}

// ---------------------------------------------------------------------------
// 3x3 conv (pad=1) as implicit GEMM on v_wmma_f32_16x16x32_bf16.
// M = C, N = B*H*W, K = 9*C (K-order (kh,kw) outer, ci inner).
// Block: 256 thr = 8 waves, 128(M) x 64(N) tile; each wave owns a 16x64 strip
// (4 WMMAs / K-step).  Register-staged software pipeline: global loads for
// tile t+1 are issued right after the barrier of tile t, so HBM/L2 latency
// hides behind the ds_load+WMMA block; the loadcnt wait lands at the next
// iteration's LDS-store site.
// ---------------------------------------------------------------------------
template <bool IS_BF16>
__global__ __launch_bounds__(256) void conv3x3_wmma(
    const void* __restrict__ xin, const unsigned short* __restrict__ wK,
    const float* __restrict__ bias, unsigned short* __restrict__ yout,
    int C, int logC, int H, int W, int B) {
  const int K = C * 9;
  const int HW = H * W;
  const int Ntot = B * HW;
  const int nBN = Ntot >> 6;
  const int m0 = (blockIdx.x / nBN) << 7;  // M tile = 128
  const int n0 = (blockIdx.x % nBN) << 6;  // N tile = 64

  __shared__ __align__(16) unsigned short As[128][32];
  __shared__ __align__(16) unsigned short Bs[64][32];

  const int t = threadIdx.x;
  const int lane = t & 31;
  const int wv = t >> 5;  // 0..7 : M sub-tile of this wave

  // A staging: 16 consecutive K per thread (two 16B loads)
  const int a_row = t >> 1;        // 0..127
  const int a_col = (t & 1) << 4;  // 0 or 16
  // B staging: 8 consecutive K (= 8 consecutive ci) of one pixel per thread
  const int b_n = t >> 2;        // 0..63
  const int b_k = (t << 3) & 31; // 0,8,16,24

  const int nglob = n0 + b_n;
  const int bb = nglob / HW;
  const int pix = nglob - bb * HW;
  const int oh = pix / W;
  const int ow = pix - oh * W;

  // fragment addressing (constant over the K loop)
  const int khalf = lane >> 4;
  const int arow = (wv << 4) + (lane & 15);
  const int kb = khalf << 4;  // lanes 0-15: K 0..15, lanes 16-31: K 16..31
  const int cl = lane & 15;

  const unsigned short* wrow = &wK[(size_t)(m0 + a_row) * K + a_col];

  v8f acc[4] = {};

  // ---- staging registers ----
  uint4 aR0, aR1;
  unsigned rawB[8];

  auto stageA = [&](int k0) {
    aR0 = *(const uint4*)(wrow + k0);
    aR1 = *(const uint4*)(wrow + k0 + 8);
    if (k0 + 32 < K) __builtin_prefetch(wrow + k0 + 32, 0, 1);
  };
  auto stageB = [&](int k0) {
    // uniform (kh,kw) for the whole K tile (C is a power of two >= 256)
    const int r = k0 >> logC;  // 0..8, scalar
    const int kh = r / 3;
    const int kw = r - kh * 3;
    const int ci0 = k0 & (C - 1);
    const int ih = oh + kh - 1;
    const int iw = ow + kw - 1;
    if (((unsigned)ih < (unsigned)H) && ((unsigned)iw < (unsigned)W)) {
      const size_t base = ((size_t)(bb * C + ci0 + b_k) * H + ih) * W + iw;
#pragma unroll
      for (int e = 0; e < 8; ++e) {
        rawB[e] = IS_BF16
                      ? (unsigned)((const unsigned short*)xin)[base + (size_t)e * HW]
                      : ((const unsigned*)xin)[base + (size_t)e * HW];
      }
    } else {
#pragma unroll
      for (int e = 0; e < 8; ++e) rawB[e] = 0u;
    }
  };

  stageA(0);
  stageB(0);

  for (int k0 = 0; k0 < K; k0 += 32) {
    __syncthreads();  // all waves done reading LDS from previous tile
    *(uint4*)&As[a_row][a_col] = aR0;
    *(uint4*)&As[a_row][a_col + 8] = aR1;
    __align__(16) unsigned short pk[8];
#pragma unroll
    for (int e = 0; e < 8; ++e) {
      if (IS_BF16) {
        pk[e] = (unsigned short)rawB[e];
      } else {
        float fv;
        const unsigned u = rawB[e];
        __builtin_memcpy(&fv, &u, 4);
        pk[e] = f2bf(fv);  // f2bf(0) == 0, so OOB zeros stay zero
      }
    }
    *(uint4*)&Bs[b_n][b_k] = *(const uint4*)pk;
    __syncthreads();  // tile visible to all waves

    // issue next tile's global loads now; latency overlaps the WMMA block
    const int kn = k0 + 32;
    if (kn < K) {
      stageA(kn);
      stageB(kn);
    }

    // ---- fragments: contiguous K order (k = 16*half + s) for A and B ----
    Frag fa;
    fa.q[0] = *(const uint4*)&As[arow][kb];         // K 0..7 / 16..23
    fa.q[1] = *(const uint4*)&As[arow][kb + 8];  // K 8..15 / 24..31
    Frag fb[4];
#pragma unroll
    for (int tn = 0; tn < 4; ++tn) {
      fb[tn].q[0] = *(const uint4*)&Bs[(tn << 4) + cl][kb];
      fb[tn].q[1] = *(const uint4*)&Bs[(tn << 4) + cl][kb + 8];
    }
#pragma unroll
    for (int tn = 0; tn < 4; ++tn) {
      acc[tn] = __builtin_amdgcn_wmma_f32_16x16x32_bf16(
          false, fa.v, false, fb[tn].v, (short)0, acc[tn], false, false);
    }
  }

  // ---- epilogue: bias + ReLU, store bf16 NCHW ----
  const int rb = (lane >> 4) << 3;
#pragma unroll
  for (int tn = 0; tn < 4; ++tn) {
    const int n = n0 + (tn << 4) + cl;
    const int b2 = n / HW;
    const int p2 = n - b2 * HW;
#pragma unroll
    for (int rr = 0; rr < 8; ++rr) {
      const int m = m0 + (wv << 4) + rb + rr;
      float v = acc[tn][rr] + bias[m];
      v = fmaxf(v, 0.f);
      yout[(size_t)(b2 * C + m) * HW + p2] = f2bf(v);
    }
  }
}

// ---------------------------------------------------------------------------
// 1x1 heads: obj = sigmoid(<x,wo>+bo), deltas = Wb x + bb; anchor decode.
// ---------------------------------------------------------------------------
__global__ void heads_kernel(const unsigned short* __restrict__ act,
                             const float* __restrict__ wo,
                             const float* __restrict__ bo,
                             const float* __restrict__ wb,
                             const float* __restrict__ bb4,
                             float* __restrict__ scores,
                             float* __restrict__ props, int C, int H, int W,
                             int B, float stride) {
  const int HW = H * W;
  const int tot = B * HW;
  int i = blockIdx.x * blockDim.x + threadIdx.x;
  if (i >= tot) return;
  const int b = i / HW;
  const int pix = i - b * HW;
  const int oh = pix / W;
  const int ow = pix - oh * W;
  float so = 0.f, d0 = 0.f, d1 = 0.f, d2 = 0.f, d3 = 0.f;
  const unsigned short* xp = act + (size_t)b * C * HW + pix;
  for (int c = 0; c < C; ++c) {
    const float v = bf2f(xp[(size_t)c * HW]);
    so += v * wo[c];
    d0 += v * wb[c];
    d1 += v * wb[C + c];
    d2 += v * wb[2 * C + c];
    d3 += v * wb[3 * C + c];
  }
  so += bo[0];
  d0 += bb4[0]; d1 += bb4[1]; d2 += bb4[2]; d3 += bb4[3];
  const float obj = 1.f / (1.f + expf(-so));
  const float cx = (ow + 0.5f) * stride + d0 * stride;
  const float cy = (oh + 0.5f) * stride + d1 * stride;
  const float pw = stride * expf(d2);
  const float ph = stride * expf(d3);
  scores[i] = obj;
  float* p = props + (size_t)i * 4;
  p[0] = cx - 0.5f * pw;
  p[1] = cy - 0.5f * ph;
  p[2] = cx + 0.5f * pw;
  p[3] = cy + 0.5f * ph;
}

// ---------------------------------------------------------------------------
// Bitonic sort machinery (descending, stable by index). Nloc = power of two,
// batches are contiguous segments of Nloc.
// ---------------------------------------------------------------------------
__global__ void init_sort(const float* __restrict__ src, float* __restrict__ key,
                          int* __restrict__ idx, int Nloc, int total) {
  int i = blockIdx.x * blockDim.x + threadIdx.x;
  if (i >= total) return;
  key[i] = src[i];
  idx[i] = i & (Nloc - 1);
}

__global__ void bitonic_stage(float* __restrict__ key, int* __restrict__ idx,
                              int Nloc, int total, int j, int k) {
  int i = blockIdx.x * blockDim.x + threadIdx.x;
  if (i >= total) return;
  int p = i ^ j;
  if (p <= i) return;  // partner is always in the same batch (j < Nloc)
  const int local = i & (Nloc - 1);
  const float ka = key[i], kb = key[p];
  const int ia = idx[i], ib = idx[p];
  const bool aGreater = (ka > kb) || (ka == kb && ia < ib);
  const bool desc = ((local & k) == 0);
  if (desc ? !aGreater : aGreater) {
    key[i] = kb; key[p] = ka;
    idx[i] = ib; idx[p] = ia;
  }
}

// gather top-Kk boxes/scores after sorting (still sorted descending)
__global__ void gather_topk(const float* __restrict__ scores,
                            const float* __restrict__ props,
                            const int* __restrict__ sidx,
                            float* __restrict__ sc3, float* __restrict__ gp,
                            int Nloc, int Kk, int B) {
  int i = blockIdx.x * blockDim.x + threadIdx.x;
  if (i >= B * Kk) return;
  const int b = i / Kk;
  const int r = i - b * Kk;
  const int src = b * Nloc + sidx[b * Nloc + r];
  sc3[i] = scores[src];
  const float* ps = props + (size_t)src * 4;
  float* pd = gp + (size_t)i * 4;
  pd[0] = ps[0]; pd[1] = ps[1]; pd[2] = ps[2]; pd[3] = ps[3];
}

// ---------------------------------------------------------------------------
// Greedy NMS, one block per batch; boxes + keep bitmask in LDS.
// ---------------------------------------------------------------------------
__global__ __launch_bounds__(256) void nms_kernel(const float* __restrict__ gp,
                                                  unsigned* __restrict__ keepm,
                                                  int Kk, float thr,
                                                  int wstride) {
  __shared__ float4 sb[3000];
  __shared__ unsigned km[(3000 + 31) / 32];
  const int b = blockIdx.x;
  const int words = (Kk + 31) / 32;
  const float4* boxes = (const float4*)(gp + (size_t)b * Kk * 4);
  for (int i = threadIdx.x; i < Kk; i += blockDim.x) sb[i] = boxes[i];
  for (int w = threadIdx.x; w < words; w += blockDim.x) km[w] = 0xFFFFFFFFu;
  __syncthreads();
  for (int i = 0; i < Kk; ++i) {
    const bool kept = (km[i >> 5] >> (i & 31)) & 1u;
    if (kept) {
      const float4 bi = sb[i];
      const float areai = (bi.z - bi.x) * (bi.w - bi.y);
      for (int j = i + 1 + (int)threadIdx.x; j < Kk; j += blockDim.x) {
        if (!((km[j >> 5] >> (j & 31)) & 1u)) continue;
        const float4 bj = sb[j];
        const float xx1 = fmaxf(bi.x, bj.x), yy1 = fmaxf(bi.y, bj.y);
        const float xx2 = fminf(bi.z, bj.z), yy2 = fminf(bi.w, bj.w);
        const float iw = fmaxf(xx2 - xx1, 0.f), ih = fmaxf(yy2 - yy1, 0.f);
        const float inter = iw * ih;
        const float areaj = (bj.z - bj.x) * (bj.w - bj.y);
        const float iou = inter / (areai + areaj - inter + 1e-9f);
        if (iou > thr) atomicAnd(&km[j >> 5], ~(1u << (j & 31)));
      }
    }
    __syncthreads();
  }
  for (int w = threadIdx.x; w < words; w += blockDim.x)
    keepm[b * wstride + w] = km[w];
}

// masked scores padded to Npad (power of two) for post-NMS top-k
__global__ void init_masked(const float* __restrict__ sc3,
                            const unsigned* __restrict__ keepm,
                            float* __restrict__ key, int* __restrict__ idx,
                            int Kk, int Npad, int wstride, int B) {
  int i = blockIdx.x * blockDim.x + threadIdx.x;
  if (i >= B * Npad) return;
  const int b = i / Npad;
  const int r = i - b * Npad;
  float v = -__builtin_inff();
  if (r < Kk) {
    const bool kept = (keepm[b * wstride + (r >> 5)] >> (r & 31)) & 1u;
    if (kept) v = sc3[b * Kk + r];
  }
  key[i] = v;
  idx[i] = r;
}

// append this level's top-kk into the concatenated per-batch pool
__global__ void append_level(const float* __restrict__ sc3,
                             const float* __restrict__ gp,
                             const int* __restrict__ sidx2,
                             float* __restrict__ catS, float* __restrict__ catP,
                             float* __restrict__ catL, int lvl, int Kk,
                             int Npad, int kk, int catStride, int B) {
  int i = blockIdx.x * blockDim.x + threadIdx.x;
  if (i >= B * kk) return;
  const int b = i / kk;
  const int r = i - b * kk;
  const int kidx = sidx2[b * Npad + r];
  const int dst = b * catStride + lvl * kk + r;
  float s = -__builtin_inff();
  float p0 = 0.f, p1 = 0.f, p2 = 0.f, p3 = 0.f;
  if (kidx < Kk) {
    const int src = b * Kk + kidx;
    s = sc3[src];  // reference takes UNMASKED scores at kidx
    const float* pp = gp + (size_t)src * 4;
    p0 = pp[0]; p1 = pp[1]; p2 = pp[2]; p3 = pp[3];
  }
  catS[dst] = s;
  float* pd = catP + (size_t)dst * 4;
  pd[0] = p0; pd[1] = p1; pd[2] = p2; pd[3] = p3;
  catL[dst] = (float)lvl;
}

__global__ void init_final(const float* __restrict__ catS,
                           float* __restrict__ key, int* __restrict__ idx,
                           int Kc, int Npad, int B) {
  int i = blockIdx.x * blockDim.x + threadIdx.x;
  if (i >= B * Npad) return;
  const int b = i / Npad;
  const int r = i - b * Npad;
  key[i] = (r < Kc) ? catS[b * Kc + r] : -__builtin_inff();
  idx[i] = r;
}

__global__ void write_out(const float* __restrict__ catS,
                          const float* __restrict__ catP,
                          const float* __restrict__ catL,
                          const int* __restrict__ sidx2,
                          float* __restrict__ out, int Kc, int Npad, int Kout,
                          int B) {
  int i = blockIdx.x * blockDim.x + threadIdx.x;
  if (i >= B * Kout) return;
  const int b = i / Kout;
  const int r = i - b * Kout;
  const int kidx = sidx2[b * Npad + r];
  float* props_out = out;                      // [B][Kout][4]
  float* sc_out = out + (size_t)B * Kout * 4;  // [B][Kout]
  float* lvl_out = sc_out + (size_t)B * Kout;  // [B][Kout]
  float s = -__builtin_inff(), lv = 0.f;
  float p0 = 0.f, p1 = 0.f, p2 = 0.f, p3 = 0.f;
  if (kidx < Kc) {
    const int src = b * Kc + kidx;
    s = catS[src];
    lv = catL[src];
    const float* pp = catP + (size_t)src * 4;
    p0 = pp[0]; p1 = pp[1]; p2 = pp[2]; p3 = pp[3];
  }
  float* pd = props_out + (size_t)(b * Kout + r) * 4;
  pd[0] = p0; pd[1] = p1; pd[2] = p2; pd[3] = p3;
  sc_out[b * Kout + r] = s;
  lvl_out[b * Kout + r] = lv;
}

// ---------------------------------------------------------------------------
// Host orchestration
// ---------------------------------------------------------------------------
extern "C" void kernel_launch(void* const* d_in, const int* in_sizes, int n_in,
                              void* d_out, int out_size, void* d_ws,
                              size_t ws_size, hipStream_t stream) {
  const int CH[3] = {256, 512, 1024};
  const int LOGC[3] = {8, 9, 10};
  const int HH[3] = {128, 64, 32};
  const float STR[3] = {8.f, 16.f, 32.f};
  const int B = in_sizes[0] / (256 * 128 * 128);
  const int TOPK = 1000;
  const float NMS_THR = 0.7f;
  (void)n_in; (void)out_size; (void)ws_size;

  char* wsb = (char*)d_ws;
  size_t cur = 0;
  auto walloc = [&](size_t bytes) -> void* {
    void* p = wsb + cur;
    cur += (bytes + 255) & ~(size_t)255;
    return p;
  };
  unsigned short* wbuf = (unsigned short*)walloc((size_t)1024 * 1024 * 9 * 2);
  unsigned short* act1 = (unsigned short*)walloc((size_t)B * 256 * 128 * 128 * 2);
  unsigned short* act2 = (unsigned short*)walloc((size_t)B * 256 * 128 * 128 * 2);
  float* scores = (float*)walloc((size_t)B * 16384 * 4);
  float* props = (float*)walloc((size_t)B * 16384 * 16);
  float* skey = (float*)walloc((size_t)B * 16384 * 4);
  int* sidx = (int*)walloc((size_t)B * 16384 * 4);
  float* sc3 = (float*)walloc((size_t)B * 3000 * 4);
  float* gp = (float*)walloc((size_t)B * 3000 * 16);
  unsigned* keepm = (unsigned*)walloc((size_t)B * 128 * 4);
  float* skey2 = (float*)walloc((size_t)B * 4096 * 4);
  int* sidx2 = (int*)walloc((size_t)B * 4096 * 4);
  float* catS = (float*)walloc((size_t)B * 3000 * 4);
  float* catP = (float*)walloc((size_t)B * 3000 * 16);
  float* catL = (float*)walloc((size_t)B * 3000 * 4);

  auto sort_desc = [&](float* key, int* idx, int Nloc, int batches) {
    const int total = Nloc * batches;
    const int blocks = (total + 255) / 256;
    for (int k = 2; k <= Nloc; k <<= 1)
      for (int j = k >> 1; j >= 1; j >>= 1)
        bitonic_stage<<<blocks, 256, 0, stream>>>(key, idx, Nloc, total, j, k);
  };

  for (int l = 0; l < 3; ++l) {
    const int C = CH[l], H = HH[l], W = H, HW = H * W, N = HW;
    const float* f = (const float*)d_in[l];
    const float* w1 = (const float*)d_in[3 + l * 8 + 0];
    const float* b1 = (const float*)d_in[3 + l * 8 + 1];
    const float* w2 = (const float*)d_in[3 + l * 8 + 2];
    const float* b2 = (const float*)d_in[3 + l * 8 + 3];
    const float* wo = (const float*)d_in[3 + l * 8 + 4];
    const float* bo = (const float*)d_in[3 + l * 8 + 5];
    const float* wb = (const float*)d_in[3 + l * 8 + 6];
    const float* bb = (const float*)d_in[3 + l * 8 + 7];

    const int wn = C * C * 9;
    const int Ntot = B * HW;
    const dim3 cgrid((C / 128) * (Ntot / 64));

    cvt_transpose_w<<<(wn + 255) / 256, 256, 0, stream>>>(w1, wbuf, C);
    conv3x3_wmma<false><<<cgrid, 256, 0, stream>>>(f, wbuf, b1, act1, C,
                                                   LOGC[l], H, W, B);
    cvt_transpose_w<<<(wn + 255) / 256, 256, 0, stream>>>(w2, wbuf, C);
    conv3x3_wmma<true><<<cgrid, 256, 0, stream>>>(act1, wbuf, b2, act2, C,
                                                  LOGC[l], H, W, B);

    heads_kernel<<<(Ntot + 255) / 256, 256, 0, stream>>>(
        act2, wo, bo, wb, bb, scores, props, C, H, W, B, STR[l]);

    // sort all scores (descending, stable) per batch; N is a power of two
    init_sort<<<(Ntot + 255) / 256, 256, 0, stream>>>(scores, skey, sidx, N,
                                                      Ntot);
    sort_desc(skey, sidx, N, B);

    const int Kk = (N < 3000) ? N : 3000;
    gather_topk<<<(B * Kk + 255) / 256, 256, 0, stream>>>(scores, props, sidx,
                                                          sc3, gp, N, Kk, B);

    nms_kernel<<<B, 256, 0, stream>>>(gp, keepm, Kk, NMS_THR, 128);

    int Np2 = 1;
    while (Np2 < Kk) Np2 <<= 1;
    init_masked<<<(B * Np2 + 255) / 256, 256, 0, stream>>>(
        sc3, keepm, skey2, sidx2, Kk, Np2, 128, B);
    sort_desc(skey2, sidx2, Np2, B);

    const int kk = (TOPK < Kk) ? TOPK : Kk;  // 1000 for all levels
    append_level<<<(B * kk + 255) / 256, 256, 0, stream>>>(
        sc3, gp, sidx2, catS, catP, catL, l, Kk, Np2, kk, 3000, B);
  }

  // final cross-level top-k
  init_final<<<(B * 4096 + 255) / 256, 256, 0, stream>>>(catS, skey2, sidx2,
                                                         3000, 4096, B);
  sort_desc(skey2, sidx2, 4096, B);
  write_out<<<(B * TOPK + 255) / 256, 256, 0, stream>>>(
      catS, catP, catL, sidx2, (float*)d_out, 3000, 4096, TOPK, B);
}
